// GCN_Net_gelu_89051851915797
// MI455X (gfx1250) — matrix operand order repaired and checked by reading, not compiled, not run on hardware
//
#include <hip/hip_runtime.h>
#include <hip/hip_bf16.h>
#include <cstdint>

// ---------------------------------------------------------------------------
// GCN link-prediction pipeline for MI455X (gfx1250, wave32, WMMA).
//
// All five GEMMs run through a bf16 WMMA kernel (v_wmma_f32_16x16x32_bf16,
// f32 accumulate) — BatchNorm follows every GEMM, so bf16 inputs are safe.
// Biases b1/b2/bl1/bl2 are mathematically absorbed by BN mean-subtraction
// and are skipped (exact identity); only bf is applied (fused in GEMM5).
// Edge aggregation = deg^{-1/2} sym-norm + f32 global atomics (bandwidth
// bound: ~2 GB irregular traffic @ 23.3 TB/s dominates runtime).
// ---------------------------------------------------------------------------

typedef __bf16 bf16;
typedef __attribute__((ext_vector_type(8)))  __bf16 bf16x8;
typedef __attribute__((ext_vector_type(16))) __bf16 bf16x16;
typedef __attribute__((ext_vector_type(8)))  float  v8f;

#define TILE_M 128
#define TILE_N 64
#define TILE_K 32
#define LDSPAD 40   // 80B rows: 16B aligned + bank-conflict-free b128 reads

// ---------------------------------------------------------------------------
// WMMA GEMM: D[M][ldD] (f32) = A[M][Kpad] (bf16) * Bt[Npad][Kpad]^T (bf16) + bias
// Bt is stored N-major (pre-transposed) so B fragments are contiguous in K.
// 256 threads = 8 waves; wave (wm,wn) owns a 32x32 tile = 2x2 WMMA tiles.
// ---------------------------------------------------------------------------
__global__ __launch_bounds__(256)
void wmma_gemm_bf16(const bf16* __restrict__ A, const bf16* __restrict__ Bt,
                    const float* __restrict__ bias, float* __restrict__ D,
                    int M, int Kpad, int Npad, int Nout, int ldD, int hasBias)
{
    __shared__ __align__(16) bf16 sA[TILE_M][LDSPAD];
    __shared__ __align__(16) bf16 sB[TILE_N][LDSPAD];

    const int tid  = threadIdx.x;
    const int lane = tid & 31;
    const int wave = tid >> 5;
    const int wm   = wave >> 1;            // 0..3 (row group)
    const int wn   = wave & 1;             // 0..1 (col group)
    const int bN0  = blockIdx.x * TILE_N;
    const int bM0  = blockIdx.y * TILE_M;

    v8f acc[2][2];
    #pragma unroll
    for (int i = 0; i < 2; ++i)
        #pragma unroll
        for (int j = 0; j < 2; ++j) {
            v8f z = {0.f,0.f,0.f,0.f,0.f,0.f,0.f,0.f};
            acc[i][j] = z;
        }

    const int arow = tid >> 1;             // 0..127
    const int acol = (tid & 1) * 16;       // 16 bf16 (32B) per thread
    const int nKt  = Kpad / TILE_K;

    for (int kt = 0; kt < nKt; ++kt) {
        const int k0 = kt * TILE_K;
        // ---- stage A tile (128x32) ----
        {
            const int gr = bM0 + arow;
            uint4* dst = (uint4*)&sA[arow][acol];
            if (gr < M) {
                const uint4* src = (const uint4*)(A + (size_t)gr * Kpad + k0 + acol);
                dst[0] = src[0]; dst[1] = src[1];
            } else {
                uint4 z{0,0,0,0}; dst[0] = z; dst[1] = z;
            }
        }
        // ---- stage B tile (64x32, N-major) ----
        if (tid < 128) {
            const int gn = bN0 + (tid >> 1);          // < Npad by grid sizing
            const uint4* src = (const uint4*)(Bt + (size_t)gn * Kpad + k0 + acol);
            uint4* dst = (uint4*)&sB[tid >> 1][acol];
            dst[0] = src[0]; dst[1] = src[1];
        }
        __syncthreads();

        // ---- fragments (ISA 7.12.2 16-bit layouts) ----
        const int ml = lane & 15, mh = lane >> 4;
        bf16x16 af[2], bf_[2];
        #pragma unroll
        for (int i = 0; i < 2; ++i) {
            const bf16* ap = &sA[wm * 32 + i * 16 + ml][0];
            bf16x8 lo = *(const bf16x8*)(ap + mh * 8);        // K = mh*8..+7
            bf16x8 hi = *(const bf16x8*)(ap + 16 + mh * 8);   // K = 16+mh*8..+7
            af[i] = __builtin_shufflevector(lo, hi, 0,1,2,3,4,5,6,7,8,9,10,11,12,13,14,15);
        }
        #pragma unroll
        for (int j = 0; j < 2; ++j) {
            const bf16* bp = &sB[wn * 32 + j * 16 + ml][0];
            bf16x8 lo = *(const bf16x8*)(bp + mh * 8);        // K = mh*8..+7
            bf16x8 hi = *(const bf16x8*)(bp + 16 + mh * 8);   // K = 16+mh*8..+7
            bf_[j] = __builtin_shufflevector(lo, hi, 0,1,2,3,4,5,6,7,8,9,10,11,12,13,14,15);
        }
        #pragma unroll
        for (int i = 0; i < 2; ++i)
            #pragma unroll
            for (int j = 0; j < 2; ++j)
                acc[i][j] = __builtin_amdgcn_wmma_f32_16x16x32_bf16(
                    false, af[i], false, bf_[j], (short)0, acc[i][j], false, false);
        __syncthreads();
    }

    // ---- store D (C/D layout: VGPR r -> M=r (+8 for upper half-wave)) ----
    const int ml = lane & 15, mh = lane >> 4;
    #pragma unroll
    for (int i = 0; i < 2; ++i)
        #pragma unroll
        for (int j = 0; j < 2; ++j) {
            const int col = bN0 + wn * 32 + j * 16 + ml;
            if (col >= Nout) continue;
            const float bv = hasBias ? bias[col] : 0.0f;
            #pragma unroll
            for (int r = 0; r < 8; ++r) {
                const int row = bM0 + wm * 32 + i * 16 + mh * 8 + r;
                if (row < M)
                    D[(size_t)row * ldD + col] = acc[i][j][r] + bv;
            }
        }
}

// ---------------------------------------------------------------------------
// Small helper kernels
// ---------------------------------------------------------------------------
__global__ void fill_f32(float* p, float v, size_t n) {
    size_t i = (size_t)blockIdx.x * blockDim.x + threadIdx.x;
    if (i < n) p[i] = v;
}

// x (f32, M x K) -> bf16 padded M x Kpad
__global__ void convert_x_bf16(const float* __restrict__ X, bf16* __restrict__ Xb,
                               int M, int K, int Kpad) {
    size_t i = (size_t)blockIdx.x * blockDim.x + threadIdx.x;
    size_t total = (size_t)M * Kpad;
    if (i >= total) return;
    int r = (int)(i / Kpad), k = (int)(i % Kpad);
    Xb[i] = (bf16)((k < K) ? X[(size_t)r * K + k] : 0.0f);
}

// W (f32, K x N row-major) -> bf16 transposed padded Npad x Kpad
__global__ void convert_wt_bf16(const float* __restrict__ W, bf16* __restrict__ Wt,
                                int K, int N, int Kpad, int Npad) {
    size_t i = (size_t)blockIdx.x * blockDim.x + threadIdx.x;
    size_t total = (size_t)Npad * Kpad;
    if (i >= total) return;
    int n = (int)(i / Kpad), k = (int)(i % Kpad);
    Wt[i] = (bf16)((n < N && k < K) ? W[(size_t)k * N + n] : 0.0f);
}

__global__ void deg_count(const int* __restrict__ dst, float* __restrict__ deg, int E) {
    int i = blockIdx.x * blockDim.x + threadIdx.x;
    if (i < E) atomicAdd(&deg[dst[i]], 1.0f);
}

__global__ void deg_to_dinv(float* d, int n) {
    int i = blockIdx.x * blockDim.x + threadIdx.x;
    if (i < n) d[i] = rsqrtf(d[i]);   // deg >= 1 (self loop)
}

// out[i][c] = h[i][c] * dinv[i]^2   (self-loop term, full overwrite)
__global__ void self_init(const float* __restrict__ h, const float* __restrict__ dinv,
                          float* __restrict__ out, int N, int H) {
    size_t i = (size_t)blockIdx.x * blockDim.x + threadIdx.x;
    if (i >= (size_t)N * H) return;
    float dv = dinv[i / H];
    out[i] = h[i] * dv * dv;
}

// out[dst] += h[src] * dinv[src]*dinv[dst]   (one block per edge)
__global__ __launch_bounds__(256)
void scatter_edges(const float* __restrict__ h, const int* __restrict__ src,
                   const int* __restrict__ dst, const float* __restrict__ dinv,
                   float* __restrict__ out, int H) {
    const int e = blockIdx.x;
    const int s = src[e], d = dst[e];
    const float norm = dinv[s] * dinv[d];
    const float* hp = h + (size_t)s * H;
    float* op = out + (size_t)d * H;
    for (int c = threadIdx.x; c < H; c += blockDim.x)
        atomicAdd(&op[c], hp[c] * norm);
}

// per-column sum & sumsq (block = H threads, grid-stride over rows)
__global__ void bn_stats(const float* __restrict__ X, float* __restrict__ stats,
                         int M, int H) {
    const int c = threadIdx.x;
    float s = 0.0f, s2 = 0.0f;
    for (int r = blockIdx.x; r < M; r += gridDim.x) {
        float v = X[(size_t)r * H + c];
        s += v; s2 += v * v;
    }
    atomicAdd(&stats[c], s);
    atomicAdd(&stats[256 + c], s2);
}

__device__ __forceinline__ float gelu_exact(float y) {
    return 0.5f * y * (1.0f + erff(y * 0.70710678118654752f));
}

// BN (training-mode batch stats) + exact GELU, emit bf16 for next GEMM
__global__ void bn_apply_gelu_bf16(const float* __restrict__ X,
                                   const float* __restrict__ stats,
                                   const float* __restrict__ g,
                                   const float* __restrict__ be,
                                   bf16* __restrict__ Y, int M, int H) {
    size_t i = (size_t)blockIdx.x * blockDim.x + threadIdx.x;
    if (i >= (size_t)M * H) return;
    int c = (int)(i % H);
    float inv = 1.0f / (float)M;
    float mu  = stats[c] * inv;
    float var = stats[256 + c] * inv - mu * mu;
    float y = (X[i] - mu) * rsqrtf(var + 1e-5f) * g[c] + be[c];
    Y[i] = (bf16)gelu_exact(y);
}

// pairs: pb[i][0:H] = h[e0[i]], pb[i][H:2H] = h[e1[i]]   (bf16 -> bf16)
__global__ __launch_bounds__(256)
void pair_gather(const bf16* __restrict__ h, const int* __restrict__ e0,
                 const int* __restrict__ e1, bf16* __restrict__ pb, int H) {
    const int i = blockIdx.x;
    const bf16* a = h + (size_t)e0[i] * H;
    const bf16* b = h + (size_t)e1[i] * H;
    bf16* o = pb + (size_t)i * 2 * H;
    for (int c = threadIdx.x; c < H; c += blockDim.x) {
        o[c]     = a[c];
        o[H + c] = b[c];
    }
}

// ---------------------------------------------------------------------------
// Host orchestration
// ---------------------------------------------------------------------------
enum { NN = 50000, NE = 500000, NP = 100000, DIN = 389, H1 = 256, H2 = 128, CC = 52,
       KP1 = 416 /* DIN padded to mult of 32 */ };

static inline void launch_gemm(const bf16* A, const bf16* Bt, const float* bias,
                               float* D, int M, int Kpad, int Npad, int Nout,
                               int ldD, hipStream_t s) {
    dim3 grid(Npad / TILE_N, (M + TILE_M - 1) / TILE_M);
    wmma_gemm_bf16<<<grid, 256, 0, s>>>(A, Bt, bias, D, M, Kpad, Npad, Nout, ldD,
                                        bias ? 1 : 0);
}

extern "C" void kernel_launch(void* const* d_in, const int* in_sizes, int n_in,
                              void* d_out, int out_size, void* d_ws, size_t ws_size,
                              hipStream_t stream) {
    const int*   edge_index = (const int*)d_in[0];
    const float* x   = (const float*)d_in[1];
    const int*   edge_id = (const int*)d_in[2];
    const float* W1  = (const float*)d_in[3];
    const float* g1  = (const float*)d_in[5];
    const float* be1 = (const float*)d_in[6];
    const float* W2  = (const float*)d_in[7];
    const float* g2  = (const float*)d_in[9];
    const float* be2 = (const float*)d_in[10];
    const float* Wl1 = (const float*)d_in[11];
    const float* g3  = (const float*)d_in[13];
    const float* be3 = (const float*)d_in[14];
    const float* Wl2 = (const float*)d_in[15];
    const float* g4  = (const float*)d_in[17];
    const float* be4 = (const float*)d_in[18];
    const float* Wf  = (const float*)d_in[19];
    const float* bfb = (const float*)d_in[20];

    const int* src = edge_index;            // [E]
    const int* dst = edge_index + NE;       // [E]
    const int* e0  = edge_id;               // [P]
    const int* e1  = edge_id + NP;          // [P]

    // ---- workspace carve-out (regions reused over pipeline lifetime) ----
    char* ws = (char*)d_ws;
    size_t off = 0;
    auto take = [&](size_t bytes) -> char* {
        char* p = ws + off;
        off += (bytes + 255) & ~(size_t)255;
        return p;
    };
    float* dinv  = (float*)take((size_t)NN * 4);
    float* stats = (float*)take(512 * 4);
    bf16* W1t  = (bf16*)take((size_t)H1 * KP1 * 2);          // 256 x 416
    bf16* W2t  = (bf16*)take((size_t)H1 * H1  * 2);          // 256 x 256
    bf16* Wl1t = (bf16*)take((size_t)H2 * (2*H1) * 2);       // 128 x 512
    bf16* Wl2t = (bf16*)take((size_t)H2 * H2  * 2);          // 128 x 128
    bf16* Wft  = (bf16*)take((size_t)64 * H2  * 2);          // 64  x 128 (C pad 64)
    // region A: xb(50000x416 bf16) -> h1b(50000x256 bf16) -> pb(100000x512 bf16)
    char* regA = take((size_t)NP * 2 * H1 * 2);              // 102.4 MB
    // region B: bufH f32 (50000x256) -> h2b bf16 -> p1b/p2b bf16
    char* regB = take((size_t)NN * H1 * 4);                  // 51.2 MB
    // region C: bufAgg f32 (50000x256) -> bufp f32 (100000x128)
    char* regC = take((size_t)NN * H1 * 4);                  // 51.2 MB

    bf16*  xb   = (bf16*)regA;
    bf16*  h1b  = (bf16*)regA;
    bf16*  pb   = (bf16*)regA;
    float* bufH = (float*)regB;
    bf16*  h2b  = (bf16*)regB;
    bf16*  p1b  = (bf16*)regB;
    bf16*  p2b  = (bf16*)regB;
    float* bufAgg = (float*)regC;
    float* bufp   = (float*)regC;

    const int TPB = 256;
    auto blks = [](size_t n, int t) { return (unsigned)((n + t - 1) / t); };

    // ---- precompute: bf16 conversions + symmetric-norm degrees ----
    convert_x_bf16<<<blks((size_t)NN * KP1, TPB), TPB, 0, stream>>>(x, xb, NN, DIN, KP1);
    convert_wt_bf16<<<blks((size_t)H1 * KP1, TPB), TPB, 0, stream>>>(W1, W1t, DIN, H1, KP1, H1);
    convert_wt_bf16<<<blks((size_t)H1 * H1, TPB), TPB, 0, stream>>>(W2, W2t, H1, H1, H1, H1);
    convert_wt_bf16<<<blks((size_t)H2 * 2 * H1, TPB), TPB, 0, stream>>>(Wl1, Wl1t, 2 * H1, H2, 2 * H1, H2);
    convert_wt_bf16<<<blks((size_t)H2 * H2, TPB), TPB, 0, stream>>>(Wl2, Wl2t, H2, H2, H2, H2);
    convert_wt_bf16<<<blks((size_t)64 * H2, TPB), TPB, 0, stream>>>(Wf, Wft, H2, CC, H2, 64);

    fill_f32<<<blks(NN, TPB), TPB, 0, stream>>>(dinv, 1.0f, NN);   // self-loop
    deg_count<<<blks(NE, TPB), TPB, 0, stream>>>(dst, dinv, NE);
    deg_to_dinv<<<blks(NN, TPB), TPB, 0, stream>>>(dinv, NN);

    // ---- layer 1: GCNConv -> BN -> GELU ----
    launch_gemm(xb, W1t, nullptr, bufH, NN, KP1, H1, H1, H1, stream);
    self_init<<<blks((size_t)NN * H1, TPB), TPB, 0, stream>>>(bufH, dinv, bufAgg, NN, H1);
    scatter_edges<<<NE, 256, 0, stream>>>(bufH, src, dst, dinv, bufAgg, H1);
    fill_f32<<<2, 256, 0, stream>>>(stats, 0.0f, 512);
    bn_stats<<<512, H1, 0, stream>>>(bufAgg, stats, NN, H1);
    bn_apply_gelu_bf16<<<blks((size_t)NN * H1, TPB), TPB, 0, stream>>>(
        bufAgg, stats, g1, be1, h1b, NN, H1);

    // ---- layer 2: GCNConv -> BN -> GELU ----
    launch_gemm(h1b, W2t, nullptr, bufH, NN, H1, H1, H1, H1, stream);
    self_init<<<blks((size_t)NN * H1, TPB), TPB, 0, stream>>>(bufH, dinv, bufAgg, NN, H1);
    scatter_edges<<<NE, 256, 0, stream>>>(bufH, src, dst, dinv, bufAgg, H1);
    fill_f32<<<2, 256, 0, stream>>>(stats, 0.0f, 512);
    bn_stats<<<512, H1, 0, stream>>>(bufAgg, stats, NN, H1);
    bn_apply_gelu_bf16<<<blks((size_t)NN * H1, TPB), TPB, 0, stream>>>(
        bufAgg, stats, g2, be2, h2b, NN, H1);

    // ---- link head: gather pairs -> 2 x (Linear->BN->GELU) -> final Linear ----
    pair_gather<<<NP, 256, 0, stream>>>(h2b, e0, e1, pb, H1);

    launch_gemm(pb, Wl1t, nullptr, bufp, NP, 2 * H1, H2, H2, H2, stream);
    fill_f32<<<2, 256, 0, stream>>>(stats, 0.0f, 512);
    bn_stats<<<512, H2, 0, stream>>>(bufp, stats, NP, H2);
    bn_apply_gelu_bf16<<<blks((size_t)NP * H2, TPB), TPB, 0, stream>>>(
        bufp, stats, g3, be3, p1b, NP, H2);

    launch_gemm(p1b, Wl2t, nullptr, bufp, NP, H2, H2, H2, H2, stream);
    fill_f32<<<2, 256, 0, stream>>>(stats, 0.0f, 512);
    bn_stats<<<512, H2, 0, stream>>>(bufp, stats, NP, H2);
    bn_apply_gelu_bf16<<<blks((size_t)NP * H2, TPB), TPB, 0, stream>>>(
        bufp, stats, g4, be4, p2b, NP, H2);

    launch_gemm(p2b, Wft, bfb, (float*)d_out, NP, H2, 64, CC, CC, stream);
}
